// GPT2_35485019800120
// MI455X (gfx1250) — matrix operand order repaired and checked by reading, not compiled, not run on hardware
//
#include <hip/hip_runtime.h>
#include <hip/hip_bf16.h>

// ---------------------------------------------------------------------------
// GPT-2 forward for MI455X (gfx1250): bf16 WMMA (f32 accum) everywhere.
// B=2, T=1024, D=768, H=12, HS=64, L=4, V=50257
// GEMM: 1 wave -> 32x64 macro-tile (8 wmma per 32-wide K-step),
//       global_prefetch of next B panel, f32->bf16 conversion in registers.
// ---------------------------------------------------------------------------

#define GB 2
#define GT 1024
#define GD 768
#define GH 12
#define GHS 64
#define GL 4
#define GV 50257
#define GM (GB * GT)          // 2048 rows
#define GD4 (4 * GD)          // 3072
#define QKVW (3 * GHS)        // 192
#define ROWQKV (GH * QKVW)    // 2304

typedef __bf16 v16bf __attribute__((ext_vector_type(16)));
typedef float  v8f   __attribute__((ext_vector_type(8)));

__device__ __forceinline__ v8f wmma_bf16(v16bf a, v16bf b, v8f c) {
  // D = A(16x32 bf16) * B(32x16 bf16) + C(16x16 f32)
  return __builtin_amdgcn_wmma_f32_16x16x32_bf16(
      /*neg_a=*/false, a, /*neg_b=*/false, b,
      /*c_mod=*/(short)0, c, /*reuse_a=*/false, /*reuse_b=*/false);
}

// A-frag (16x32 bf16): lane -> m (lane&15); elem e -> k = e + 16*half
// p must point at row + k0 + 16*half (16-byte aligned).
__device__ __forceinline__ v16bf load_a_frag(const float* p) {
  const float4* ap = reinterpret_cast<const float4*>(p);
  float4 a0 = ap[0], a1 = ap[1], a2 = ap[2], a3 = ap[3];
  v16bf a;
  a[0]=(__bf16)a0.x;  a[1]=(__bf16)a0.y;  a[2]=(__bf16)a0.z;  a[3]=(__bf16)a0.w;
  a[4]=(__bf16)a1.x;  a[5]=(__bf16)a1.y;  a[6]=(__bf16)a1.z;  a[7]=(__bf16)a1.w;
  a[8]=(__bf16)a2.x;  a[9]=(__bf16)a2.y;  a[10]=(__bf16)a2.z; a[11]=(__bf16)a2.w;
  a[12]=(__bf16)a3.x; a[13]=(__bf16)a3.y; a[14]=(__bf16)a3.z; a[15]=(__bf16)a3.w;
  return a;
}

// B-frag (32x16 bf16): lane -> n (lane&15); elem e -> k = e + 16*half
// bp must point at B + (k0 + 16*half)*ldb + col.
__device__ __forceinline__ v16bf load_b_frag(const float* bp, size_t ldb) {
  v16bf b;
#pragma unroll
  for (int e = 0; e < 16; ++e) b[e] = (__bf16)bp[(size_t)e * ldb];
  return b;
}

enum { GEMM_ADD = 1, GEMM_RELU = 2 };

// ---------------------------------------------------------------------------
// Embedding: x[b,t,:] = emb[idx[b,t],:] + pos[t,:]
// ---------------------------------------------------------------------------
__global__ void embed_kernel(const int* __restrict__ idx,
                             const float* __restrict__ emb,
                             const float* __restrict__ pos,
                             float* __restrict__ x) {
  size_t i = (size_t)blockIdx.x * blockDim.x + threadIdx.x;
  if (i >= (size_t)GM * GD) return;
  int d   = (int)(i % GD);
  int row = (int)(i / GD);          // b*T + t
  int t   = row % GT;
  int tok = idx[row];
  x[i] = emb[(size_t)tok * GD + d] + pos[(size_t)t * GD + d];
}

// ---------------------------------------------------------------------------
// LayerNorm over D=768, one 256-thread block per row.
// ---------------------------------------------------------------------------
__global__ void ln_kernel(const float* __restrict__ x,
                          const float* __restrict__ g,
                          const float* __restrict__ beta,
                          float* __restrict__ out) {
  __shared__ float red[256];
  int row = blockIdx.x;
  int tid = threadIdx.x;
  const float* xr = x + (size_t)row * GD;
  float v0 = xr[tid], v1 = xr[tid + 256], v2 = xr[tid + 512];
  red[tid] = v0 + v1 + v2;
  __syncthreads();
  for (int s = 128; s > 0; s >>= 1) {
    if (tid < s) red[tid] += red[tid + s];
    __syncthreads();
  }
  float mu = red[0] * (1.0f / GD);
  __syncthreads();
  float d0 = v0 - mu, d1 = v1 - mu, d2 = v2 - mu;
  red[tid] = d0 * d0 + d1 * d1 + d2 * d2;
  __syncthreads();
  for (int s = 128; s > 0; s >>= 1) {
    if (tid < s) red[tid] += red[tid + s];
    __syncthreads();
  }
  float rstd = rsqrtf(red[0] * (1.0f / GD) + 1e-5f);
  float* orow = out + (size_t)row * GD;
  orow[tid]       = d0 * rstd * g[tid]       + beta[tid];
  orow[tid + 256] = d1 * rstd * g[tid + 256] + beta[tid + 256];
  orow[tid + 512] = d2 * rstd * g[tid + 512] + beta[tid + 512];
}

// ---------------------------------------------------------------------------
// WMMA GEMM: one wave computes a 32(M) x 64(N) macro-tile of
//   C[M,N] (op)= A[M,K] @ B[K,N] + bias
// 8 wmma per 32-wide K-step (2 A-frags x 4 B-frags). K % 32 == 0.
// N may be ragged (LM head): loads clamped, stores masked. Garbage columns
// only pollute their own lane's output column, which is masked on store.
// C/D layout (16x16 f32): lane -> n (lane&15); vgpr r -> m = r + 8*half.
// ---------------------------------------------------------------------------
template <int FLAGS>
__global__ void gemm32x64(const float* __restrict__ A, int lda,
                          const float* __restrict__ B, int ldb,
                          const float* __restrict__ bias,
                          float* __restrict__ C, int ldc,
                          int N, int K) {
  const int lane = threadIdx.x;
  const int lm   = lane & 15;
  const int half = lane >> 4;
  const int m0   = blockIdx.y * 32;
  const int n0   = blockIdx.x * 64;

  int ncl[4];
#pragma unroll
  for (int t = 0; t < 4; ++t) {
    int n = n0 + 16 * t + lm;
    ncl[t] = (n < N) ? n : (N - 1);   // clamp so loads stay in-bounds
  }

  v8f acc[2][4];
#pragma unroll
  for (int s = 0; s < 2; ++s)
#pragma unroll
    for (int t = 0; t < 4; ++t) acc[s][t] = (v8f){};

  const float* arow0 = A + (size_t)(m0 + lm) * lda;
  const float* arow1 = arow0 + (size_t)16 * lda;

  for (int k0 = 0; k0 < K; k0 += 32) {
    // Prefetch next K-step's B panel (speculative; dropped if past the end).
    if (k0 + 32 < K) {
      const char* pf = (const char*)(B + (size_t)(k0 + 32 + lane) * ldb + n0);
      __builtin_prefetch(pf, 0, 1);
      __builtin_prefetch(pf + 128, 0, 1);
    }

    v16bf a0 = load_a_frag(arow0 + k0 + 16 * half);
    v16bf a1 = load_a_frag(arow1 + k0 + 16 * half);
    const float* bbase = B + (size_t)(k0 + 16 * half) * ldb;
#pragma unroll
    for (int t = 0; t < 4; ++t) {
      v16bf bf = load_b_frag(bbase + ncl[t], (size_t)ldb);
      acc[0][t] = wmma_bf16(a0, bf, acc[0][t]);
      acc[1][t] = wmma_bf16(a1, bf, acc[1][t]);
    }
  }

#pragma unroll
  for (int t = 0; t < 4; ++t) {
    int n = n0 + 16 * t + lm;
    if (n < N) {
      float bv = bias ? bias[n] : 0.0f;
#pragma unroll
      for (int s = 0; s < 2; ++s) {
#pragma unroll
        for (int r = 0; r < 8; ++r) {
          int m = m0 + 16 * s + r + 8 * half;
          size_t off = (size_t)m * ldc + n;
          float v = acc[s][t][r] + bv;
          if (FLAGS & GEMM_RELU) v = fmaxf(v, 0.0f);
          if (FLAGS & GEMM_ADD)  v += C[off];
          C[off] = v;
        }
      }
    }
  }
}

// ---------------------------------------------------------------------------
// Flash-style causal attention. qkv layout: [B, T, H, 192] (q|k|v per head).
// One wave per (b, h, 16-query tile); 32 keys per iteration.
// Output o: [B, T, H*HS]
// ---------------------------------------------------------------------------
__global__ void attn_kernel(const float* __restrict__ qkv,
                            float* __restrict__ o) {
  const int lane = threadIdx.x;
  const int lm   = lane & 15;
  const int half = lane >> 4;
  const int qt = blockIdx.x, h = blockIdx.y, b = blockIdx.z;
  const int qbase = qt * 16;
  const size_t rs = (size_t)ROWQKV;   // 2304 floats per (b,t) row
  const float* base = qkv + (size_t)b * GT * rs + (size_t)h * QKVW;

  // Q A-fragments for the two 32-wide K-steps (HS=64), pre-scaled by 1/sqrt(HS)
  v16bf aq[2];
  {
    const float* qrow = base + (size_t)(qbase + lm) * rs;   // q at offset 0
#pragma unroll
    for (int s = 0; s < 2; ++s) {
      const float* p = qrow + 32 * s + 16 * half;
#pragma unroll
      for (int e = 0; e < 8; ++e) {
        aq[s][e]     = (__bf16)(p[e]      * 0.125f);
        aq[s][e + 8] = (__bf16)(p[e + 8] * 0.125f);
      }
    }
  }

  float mrow[8], lrow[8];
#pragma unroll
  for (int r = 0; r < 8; ++r) { mrow[r] = -1e30f; lrow[r] = 0.0f; }
  v8f acc0 = {}, acc1 = {}, acc2 = {}, acc3 = {};

  __shared__ float ldsP[16 * 32];

  for (int j = 0; j < qbase + 16; j += 32) {
    // --- S = (Q/sqrt(d)) @ K^T, two 16x16 key tiles ---
    v8f S0 = {}, S1 = {};
#pragma unroll
    for (int s = 0; s < 2; ++s) {
      const float* k0p = base + (size_t)(j + lm) * rs + GHS + 32 * s + 16 * half;
      const float* k1p = k0p + 16 * rs;
      v16bf bk0, bk1;
#pragma unroll
      for (int e = 0; e < 16; ++e) {
        bk0[e] = (__bf16)k0p[e];
        bk1[e] = (__bf16)k1p[e];
      }
      S0 = wmma_bf16(aq[s], bk0, S0);
      S1 = wmma_bf16(aq[s], bk1, S1);
    }
    // --- causal mask (only when tile straddles the diagonal) ---
    if (j + 31 > qbase) {
#pragma unroll
      for (int r = 0; r < 8; ++r) {
        int q = qbase + r + 8 * half;
        if (j + lm > q)      S0[r] = -1e30f;
        if (j + 16 + lm > q) S1[r] = -1e30f;
      }
    }
    // --- online softmax: rows live entirely in one 16-lane half-group ---
#pragma unroll
    for (int r = 0; r < 8; ++r) {
      float mx = fmaxf(S0[r], S1[r]);
#pragma unroll
      for (int off = 8; off >= 1; off >>= 1)
        mx = fmaxf(mx, __shfl_xor(mx, off, 32));
      float mnew = fmaxf(mrow[r], mx);
      float corr = __expf(mrow[r] - mnew);
      float p0 = __expf(S0[r] - mnew);
      float p1 = __expf(S1[r] - mnew);
      float ps = p0 + p1;
#pragma unroll
      for (int off = 8; off >= 1; off >>= 1)
        ps += __shfl_xor(ps, off, 32);
      lrow[r] = lrow[r] * corr + ps;
      mrow[r] = mnew;
      acc0[r] *= corr; acc1[r] *= corr; acc2[r] *= corr; acc3[r] *= corr;
      S0[r] = p0; S1[r] = p1;
    }
    // --- transpose P (C-layout) -> A-layout via LDS ---
#pragma unroll
    for (int r = 0; r < 8; ++r) {
      int m = r + 8 * half;
      ldsP[m * 32 + lm]      = S0[r];
      ldsP[m * 32 + 16 + lm] = S1[r];
    }
    __syncthreads();
    v16bf aP;
    {
      const float* pp = &ldsP[lm * 32 + 16 * half];
#pragma unroll
      for (int e = 0; e < 8; ++e) {
        aP[e]     = (__bf16)pp[e];
        aP[e + 8] = (__bf16)pp[e + 8];
      }
    }
    __syncthreads();
    // --- O += P @ V (four 16-wide HS tiles) ---
    const float* vbase = base + (size_t)(j + 16 * half) * rs + 2 * GHS + lm;
    v16bf bv0, bv1, bv2, bv3;
#pragma unroll
    for (int e = 0; e < 16; ++e) {
      const float* vp = vbase + (size_t)e * rs;
      bv0[e] = (__bf16)vp[0];
      bv1[e] = (__bf16)vp[16];
      bv2[e] = (__bf16)vp[32];
      bv3[e] = (__bf16)vp[48];
    }
    acc0 = wmma_bf16(aP, bv0, acc0);
    acc1 = wmma_bf16(aP, bv1, acc1);
    acc2 = wmma_bf16(aP, bv2, acc2);
    acc3 = wmma_bf16(aP, bv3, acc3);
  }

  // --- epilogue: O / l, write o[b, t, h*64 + n] ---
#pragma unroll
  for (int r = 0; r < 8; ++r) {
    int m = qbase + r + 8 * half;
    float inv = 1.0f / lrow[r];
    float* orow = o + ((size_t)(b * GT + m)) * GD + h * GHS + lm;
    orow[0]  = acc0[r] * inv;
    orow[16] = acc1[r] * inv;
    orow[32] = acc2[r] * inv;
    orow[48] = acc3[r] * inv;
  }
}

// ---------------------------------------------------------------------------
// Host orchestration
// ---------------------------------------------------------------------------
extern "C" void kernel_launch(void* const* d_in, const int* in_sizes, int n_in,
                              void* d_out, int out_size, void* d_ws, size_t ws_size,
                              hipStream_t stream) {
  (void)in_sizes; (void)n_in; (void)out_size; (void)ws_size;
  const int*   idx   = (const int*)d_in[0];
  const float* emb   = (const float*)d_in[1];
  const float* pos   = (const float*)d_in[2];
  const float* Wqkv  = (const float*)d_in[3];
  const float* Wproj = (const float*)d_in[4];
  const float* bproj = (const float*)d_in[5];
  const float* W1    = (const float*)d_in[6];
  const float* b1    = (const float*)d_in[7];
  const float* W2    = (const float*)d_in[8];
  const float* b2    = (const float*)d_in[9];
  const float* g1    = (const float*)d_in[10];
  const float* beta1 = (const float*)d_in[11];
  const float* gf    = (const float*)d_in[12];
  const float* bf    = (const float*)d_in[13];
  const float* Wlm   = (const float*)d_in[14];
  const float* blm   = (const float*)d_in[15];
  float* out = (float*)d_out;

  // Workspace layout (floats): x, h, qkv, o, mid, xf  ~69 MB total
  float* ws   = (float*)d_ws;
  float* x    = ws;
  float* h    = x    + (size_t)GM * GD;
  float* qkvb = h    + (size_t)GM * GD;
  float* o    = qkvb + (size_t)GM * ROWQKV;
  float* mid  = o    + (size_t)GM * GD;
  float* xf   = mid  + (size_t)GM * GD4;

  // 1) embedding
  {
    size_t n = (size_t)GM * GD;
    embed_kernel<<<dim3((unsigned)((n + 255) / 256)), dim3(256), 0, stream>>>(
        idx, emb, pos, x);
  }

  const dim3 wave(32);
  for (int l = 0; l < GL; ++l) {
    // 2) h = LN(x)
    ln_kernel<<<dim3(GM), dim3(256), 0, stream>>>(
        x, g1 + (size_t)l * GD, beta1 + (size_t)l * GD, h);

    // 3) qkv[b,t,h,:] = h @ Wqkv[l,h]   (one GEMM per head, N=192=3x64)
    for (int hd = 0; hd < GH; ++hd) {
      const float* Bh = Wqkv + (((size_t)l * GH + hd) * GD) * QKVW;
      gemm32x64<0><<<dim3(QKVW / 64, GM / 32), wave, 0, stream>>>(
          h, GD, Bh, QKVW, nullptr, qkvb + hd * QKVW, ROWQKV, QKVW, GD);
    }

    // 4) causal attention -> o[b,t,H*HS]
    attn_kernel<<<dim3(GT / 16, GH, GB), wave, 0, stream>>>(qkvb, o);

    // 5) x += o @ Wproj[l] + bproj[l]
    gemm32x64<GEMM_ADD><<<dim3(GD / 64, GM / 32), wave, 0, stream>>>(
        o, GD, Wproj + (size_t)l * GD * GD, GD,
        bproj + (size_t)l * GD, x, GD, GD, GD);

    // 6) mid = relu(h @ W1[l] + b1[l])
    gemm32x64<GEMM_RELU><<<dim3(GD4 / 64, GM / 32), wave, 0, stream>>>(
        h, GD, W1 + (size_t)l * GD * GD4, GD4,
        b1 + (size_t)l * GD4, mid, GD4, GD4, GD);

    // 7) x += mid @ W2[l] + b2[l]
    gemm32x64<GEMM_ADD><<<dim3(GD / 64, GM / 32), wave, 0, stream>>>(
        mid, GD4, W2 + (size_t)l * GD4 * GD, GD,
        b2 + (size_t)l * GD, x, GD, GD, GD4);
  }

  // 8) xf = LN(x)
  ln_kernel<<<dim3(GM), dim3(256), 0, stream>>>(x, gf, bf, xf);

  // 9) logits = xf @ Wlm + blm   (ragged N = 50257)
  gemm32x64<0><<<dim3((GV + 63) / 64, GM / 32), wave, 0, stream>>>(
      xf, GD, Wlm, GV, blm, out, GV, GV, GD);
}
